// VisionTransformer_42296837931590
// MI455X (gfx1250) — matrix operand order repaired and checked by reading, not compiled, not run on hardware
//
#include <hip/hip_runtime.h>
#include <hip/hip_bf16.h>
#include <cstdint>

// ---------------- ViT-Base/16 forward, B=32, fp16 WMMA compute, f32 accumulate ----------------
#define IMG   224
#define PATCH 16
#define CH    3
#define DIM   768
#define DEPTH 12
#define HEADS 12
#define DK    64
#define MLPD  3072
#define NCLS  1000
#define BATCH 32
#define GRIDP 14
#define NPAT  196          // 14*14
#define SEQ   197          // tokens
#define SP    224          // padded seq for score rows (multiple of 32)
#define MTOK  (BATCH*SEQ)  // 6304 token rows
#define MPAT  (BATCH*NPAT) // 6272 patch rows

typedef _Float16 h8v  __attribute__((ext_vector_type(8)));
typedef _Float16 v16h __attribute__((ext_vector_type(16)));
typedef float    v8f  __attribute__((ext_vector_type(8)));

// ---------------------------------------------------------------------------
__global__ __launch_bounds__(256) void convert_f32_f16(const float* __restrict__ src,
                                                       _Float16* __restrict__ dst, long long n) {
  long long i = (long long)blockIdx.x * blockDim.x + threadIdx.x;
  long long stride = (long long)gridDim.x * blockDim.x;
  for (; i < n; i += stride) dst[i] = (_Float16)src[i];
}

// ---------------------------------------------------------------------------
// Image -> patch matrix [B*196, 768] f16: row = b*196 + gy*14 + gx ; col = c*256 + py*16 + px
__global__ __launch_bounds__(256) void patchify(const float* __restrict__ x,
                                                _Float16* __restrict__ xp) {
  const long long total = (long long)MPAT * DIM;
  long long i = (long long)blockIdx.x * blockDim.x + threadIdx.x;
  long long stride = (long long)gridDim.x * blockDim.x;
  for (; i < total; i += stride) {
    int row = (int)(i / DIM), pd = (int)(i % DIM);
    int b = row / NPAT, n = row % NPAT;
    int gy = n / GRIDP, gx = n % GRIDP;
    int c = pd / (PATCH * PATCH), r = pd % (PATCH * PATCH);
    int py = r / PATCH, px = r % PATCH;
    long long src = (((long long)(b * CH + c) * IMG) + gy * PATCH + py) * IMG + gx * PATCH + px;
    xp[i] = (_Float16)x[src];
  }
}

// ---------------------------------------------------------------------------
__global__ __launch_bounds__(256) void add_cls_pos(const _Float16* __restrict__ e16,
                                                   const float* __restrict__ cls,
                                                   const float* __restrict__ pos,
                                                   float* __restrict__ h) {
  const long long total = (long long)MTOK * DIM;
  long long i = (long long)blockIdx.x * blockDim.x + threadIdx.x;
  long long stride = (long long)gridDim.x * blockDim.x;
  for (; i < total; i += stride) {
    int bs = (int)(i / DIM), d = (int)(i % DIM);
    int b = bs / SEQ, s = bs % SEQ;
    float v = (s == 0) ? cls[d] : (float)e16[((long long)(b * NPAT + s - 1)) * DIM + d];
    h[i] = v + pos[(long long)s * DIM + d];
  }
}

// ---------------------------------------------------------------------------
// Row LayerNorm over D=768 -> f16. One block (256 thr) per row.
__global__ __launch_bounds__(256) void layernorm_rows(const float* __restrict__ x,
                                                      const float* __restrict__ g,
                                                      const float* __restrict__ b,
                                                      _Float16* __restrict__ y) {
  __shared__ float red[256];
  const int tid = threadIdx.x;
  const long long base = (long long)blockIdx.x * DIM;
  float v0 = x[base + tid], v1 = x[base + tid + 256], v2 = x[base + tid + 512];
  red[tid] = v0 + v1 + v2;
  __syncthreads();
  for (int o = 128; o > 0; o >>= 1) { if (tid < o) red[tid] += red[tid + o]; __syncthreads(); }
  float mu = red[0] * (1.0f / DIM);
  __syncthreads();
  float d0 = v0 - mu, d1 = v1 - mu, d2 = v2 - mu;
  red[tid] = d0 * d0 + d1 * d1 + d2 * d2;
  __syncthreads();
  for (int o = 128; o > 0; o >>= 1) { if (tid < o) red[tid] += red[tid + o]; __syncthreads(); }
  float rs = rsqrtf(red[0] * (1.0f / DIM) + 1e-5f);
  y[base + tid]       = (_Float16)(d0 * rs * g[tid]       + b[tid]);
  y[base + tid + 256] = (_Float16)(d1 * rs * g[tid + 256] + b[tid + 256]);
  y[base + tid + 512] = (_Float16)(d2 * rs * g[tid + 512] + b[tid + 512]);
}

// ---------------------------------------------------------------------------
// Softmax over one score row (197 valid of 224); writes f16 probs, zero-pads tail.
__global__ __launch_bounds__(256) void softmax_rows(const float* __restrict__ sc,
                                                    _Float16* __restrict__ p) {
  __shared__ float red[256];
  const int tid = threadIdx.x;
  const long long base = (long long)blockIdx.x * SP;
  const bool valid = (tid < SEQ);
  float v = valid ? sc[base + tid] : -3.4e38f;
  red[tid] = v;
  __syncthreads();
  for (int o = 128; o > 0; o >>= 1) { if (tid < o) red[tid] = fmaxf(red[tid], red[tid + o]); __syncthreads(); }
  float m = red[0];
  __syncthreads();
  float e = valid ? expf(v - m) : 0.0f;
  red[tid] = e;
  __syncthreads();
  for (int o = 128; o > 0; o >>= 1) { if (tid < o) red[tid] += red[tid + o]; __syncthreads(); }
  float inv = 1.0f / red[0];
  if (tid < SP) p[base + tid] = (_Float16)(valid ? e * inv : 0.0f);
}

// ---------------------------------------------------------------------------
// Tiled WMMA GEMM:  C[z] = alpha * A[z] * B[z] (+bias) -> f16 out and/or f32 out/accum
//   Block tile 128x64, K-step 32, 8 waves (256 thr) in 4(M)x2(N) grid.
//   Each wave computes a 32x32 slab: 2 A-frags x 2 B-frags = 4 WMMAs / K-step.
//   Branchless staging:
//    - A rows clamped to M-1 (masked rows only feed masked outputs; clamped reads in-bounds).
//    - B rows >= K zeroed via uniform vector select (0 * Inf = NaN otherwise).
//    - All clamped/over-reads stay inside the workspace by construction.
//   bT: B given as [N,K] row-major (multiply by B^T); else [K,N] row-major.
//   Batch offset: (z/zdiv)*Out + (z%zdiv)*In  for A, B and C.
template <int GELU, int RES>
__global__ __launch_bounds__(256) void gemm_wmma_f16(
    const _Float16* __restrict__ A, const _Float16* __restrict__ Bm,
    const float* __restrict__ bias,
    _Float16* __restrict__ outH, float* __restrict__ outF,
    int M, int N, int K, int lda, int ldb, int ldc,
    float alpha, int bT, int zdiv,
    long long aOut, long long aIn, long long bOut, long long bIn,
    long long cOut, long long cIn) {
  constexpr int LP = 40;                 // padded LDS K-stride (halves): conflict-free, 16B aligned
  __shared__ _Float16 As[128 * LP];
  __shared__ _Float16 Bs[64 * LP];

  const int tid  = threadIdx.x;
  const int lane = tid & 31;
  const int wid  = tid >> 5;
  const int wm   = wid >> 1;             // 4 waves along M (32 rows each)
  const int wn   = wid & 1;              // 2 waves along N (32 cols each)
  const int m0   = blockIdx.y * 128;
  const int n0   = blockIdx.x * 64;

  const int z  = blockIdx.z;
  const int zo = z / zdiv, zi = z % zdiv;
  A  += (long long)zo * aOut + (long long)zi * aIn;
  Bm += (long long)zo * bOut + (long long)zi * bIn;
  const long long cb = (long long)zo * cOut + (long long)zi * cIn;

  v8f acc00 = {}, acc01 = {}, acc10 = {}, acc11 = {};
  const h8v hz = {};

  for (int k0 = 0; k0 < K; k0 += 32) {
    // ---- stage A tile 128x32: 16 halves/thread, rows clamped (branchless) ----
#pragma unroll
    for (int j = 0; j < 2; ++j) {
      const int idx = tid + j * 256;
      const int row = idx >> 2, c8 = (idx & 3) * 8;
      int gm = m0 + row; gm = (gm < M) ? gm : (M - 1);
      const _Float16* ap = A + (long long)gm * lda + (k0 + c8);
      h8v v = *(const h8v*)ap;
      __builtin_prefetch(ap + 32, 0, 1);
      *(h8v*)&As[row * LP + c8] = v;
    }
    // ---- stage B tile into Bs[n][k] (K fast) ----
    if (bT) {  // B given as [N,K]: contiguous along K; clamp n row, zero k >= K per element
      const int row = tid >> 2, c8 = (tid & 3) * 8;
      int gn = n0 + row; gn = (gn < N) ? gn : (N - 1);
      const int gk = k0 + c8;
      h8v v = *(const h8v*)(Bm + (long long)gn * ldb + gk);
#pragma unroll
      for (int i = 0; i < 8; ++i) v[i] = (gk + i < K) ? v[i] : (_Float16)0;
      *(h8v*)&Bs[row * LP + c8] = v;
    } else {   // B given as [K,N]: coalesced along N; clamp k row, zero whole vector if k >= K
      const int kr = tid >> 3, n8 = (tid & 7) * 8;
      const int gk = k0 + kr;
      const int gkc = (gk < K) ? gk : (K - 1);
      h8v v = *(const h8v*)(Bm + (long long)gkc * ldb + (n0 + n8));
      v = (gk < K) ? v : hz;
#pragma unroll
      for (int i = 0; i < 8; ++i) Bs[(n8 + i) * LP + kr] = v[i];
    }
    __syncthreads();

    // ---- fragments ----
    const int l15 = lane & 15;
    const int akb = (lane < 16) ? 0 : 8;     // A: lanes 0-15 K{0-7,16-23}; 16-31 K{8-15,24-31}
    const int bkb = (lane < 16) ? 0 : 8;     // B: lanes 0-15 K{0-7,16-23}; 16-31 K{8-15,24-31}
    const int ar0 = wm * 32 + l15;
    const int ar1 = ar0 + 16;
    h8v a0lo = *(const h8v*)&As[ar0 * LP + akb];
    h8v a0hi = *(const h8v*)&As[ar0 * LP + akb + 16];
    v16h af0 = __builtin_shufflevector(a0lo, a0hi, 0,1,2,3,4,5,6,7,8,9,10,11,12,13,14,15);
    h8v a1lo = *(const h8v*)&As[ar1 * LP + akb];
    h8v a1hi = *(const h8v*)&As[ar1 * LP + akb + 16];
    v16h af1 = __builtin_shufflevector(a1lo, a1hi, 0,1,2,3,4,5,6,7,8,9,10,11,12,13,14,15);

    const int bc0 = wn * 32 + l15;
    const int bc1 = bc0 + 16;
    h8v b0lo = *(const h8v*)&Bs[bc0 * LP + bkb];
    h8v b0hi = *(const h8v*)&Bs[bc0 * LP + bkb + 16];
    v16h bf0 = __builtin_shufflevector(b0lo, b0hi, 0,1,2,3,4,5,6,7,8,9,10,11,12,13,14,15);
    h8v b1lo = *(const h8v*)&Bs[bc1 * LP + bkb];
    h8v b1hi = *(const h8v*)&Bs[bc1 * LP + bkb + 16];
    v16h bf1 = __builtin_shufflevector(b1lo, b1hi, 0,1,2,3,4,5,6,7,8,9,10,11,12,13,14,15);

    acc00 = __builtin_amdgcn_wmma_f32_16x16x32_f16(false, af0, false, bf0, (short)0, acc00, false, false);
    acc01 = __builtin_amdgcn_wmma_f32_16x16x32_f16(false, af0, false, bf1, (short)0, acc01, false, false);
    acc10 = __builtin_amdgcn_wmma_f32_16x16x32_f16(false, af1, false, bf0, (short)0, acc10, false, false);
    acc11 = __builtin_amdgcn_wmma_f32_16x16x32_f16(false, af1, false, bf1, (short)0, acc11, false, false);
    __syncthreads();
  }

  // ---- epilogue: VGPR r, lanes 0-15 -> M=r, lanes 16-31 -> M=8+r; N = lane&15
  const int mbase = m0 + wm * 32 + ((lane >> 4) << 3);
  const int nbase = n0 + wn * 32 + (lane & 15);
#pragma unroll
  for (int am = 0; am < 2; ++am) {
#pragma unroll
    for (int t = 0; t < 2; ++t) {
      v8f a = am ? (t ? acc11 : acc10) : (t ? acc01 : acc00);
      const int n = nbase + t * 16;
      if (n < N) {
        const float bb = bias ? bias[n] : 0.0f;
#pragma unroll
        for (int r = 0; r < 8; ++r) {
          const int m = mbase + am * 16 + r;
          if (m < M) {
            float gval = a[r] * alpha + bb;
            if (GELU) gval = 0.5f * gval * (1.0f + erff(gval * 0.70710678118654752f));
            const long long off = cb + (long long)m * ldc + n;
            if (outH) outH[off] = (_Float16)gval;
            if (outF) { if (RES) outF[off] += gval; else outF[off] = gval; }
          }
        }
      }
    }
  }
}

// ---------------------------------------------------------------------------
extern "C" void kernel_launch(void* const* d_in, const int* in_sizes, int n_in,
                              void* d_out, int out_size, void* d_ws, size_t ws_size,
                              hipStream_t stream) {
  (void)in_sizes; (void)n_in; (void)out_size; (void)ws_size;
  const float* x       = (const float*)d_in[0];
  const float* patchW  = (const float*)d_in[1];
  const float* patchB  = (const float*)d_in[2];
  const float* clsTok  = (const float*)d_in[3];
  const float* posEmb  = (const float*)d_in[4];
  const float* Wq = (const float*)d_in[5],  *bq = (const float*)d_in[6];
  const float* Wk = (const float*)d_in[7],  *bk = (const float*)d_in[8];
  const float* Wv = (const float*)d_in[9],  *bv = (const float*)d_in[10];
  const float* Wo = (const float*)d_in[11], *bo = (const float*)d_in[12];
  const float* ln1g = (const float*)d_in[13], *ln1b = (const float*)d_in[14];
  const float* ln2g = (const float*)d_in[15], *ln2b = (const float*)d_in[16];
  const float* W1 = (const float*)d_in[17], *b1 = (const float*)d_in[18];
  const float* W2 = (const float*)d_in[19], *b2 = (const float*)d_in[20];
  const float* lnpg = (const float*)d_in[21], *lnpb = (const float*)d_in[22];
  const float* headW = (const float*)d_in[23], *headB = (const float*)d_in[24];
  float* out = (float*)d_out;

  // ---- workspace carve (256B aligned) ----
  size_t off = 0;
  auto alloc = [&](size_t bytes) -> void* {
    off = (off + 255) & ~(size_t)255;
    void* p = (char*)d_ws + off;
    off += bytes;
    return p;
  };
  const long long szDD  = (long long)DIM * DIM;            // 589824
  const long long szDM  = (long long)DIM * MLPD;           // 2359296
  _Float16* wPatch16 = (_Float16*)alloc(szDD * 2);
  _Float16* wQ16 = (_Float16*)alloc(DEPTH * szDD * 2);
  _Float16* wK16 = (_Float16*)alloc(DEPTH * szDD * 2);
  _Float16* wV16 = (_Float16*)alloc(DEPTH * szDD * 2);
  _Float16* wO16 = (_Float16*)alloc(DEPTH * szDD * 2);
  _Float16* w116 = (_Float16*)alloc(DEPTH * szDM * 2);
  _Float16* w216 = (_Float16*)alloc(DEPTH * szDM * 2);
  _Float16* wH16 = (_Float16*)alloc((long long)DIM * NCLS * 2);
  _Float16* xp   = (_Float16*)alloc((long long)MPAT * DIM * 2);
  _Float16* e16  = (_Float16*)alloc((long long)MPAT * DIM * 2);
  float*    h32  = (float*)   alloc((long long)MTOK * DIM * 4);
  _Float16* y16  = (_Float16*)alloc((long long)MTOK * DIM * 2);
  _Float16* q16  = (_Float16*)alloc((long long)MTOK * DIM * 2);
  _Float16* k16  = (_Float16*)alloc((long long)MTOK * DIM * 2);
  _Float16* v16  = (_Float16*)alloc((long long)MTOK * DIM * 2);
  float*    sc32 = (float*)   alloc((long long)BATCH * HEADS * SEQ * SP * 4);
  _Float16* p16  = (_Float16*)alloc((long long)BATCH * HEADS * SEQ * SP * 2);
  _Float16* att  = (_Float16*)alloc((long long)MTOK * DIM * 2);
  _Float16* mlp  = (_Float16*)alloc((long long)MTOK * MLPD * 2);
  (void)alloc(4096);  // tail guard so clamped over-reads of the last buffer stay in ws

  // ---- generic GEMM dispatch ----
  auto gemm = [&](const _Float16* A, const _Float16* Bm, const float* bias,
                  _Float16* oh, float* of,
                  int M, int N, int K, int lda, int ldb, int ldc,
                  float alpha, int bT, int batch, int zdiv,
                  long long aOut, long long aIn, long long bOut, long long bIn,
                  long long cOut, long long cIn, int gelu, int res) {
    dim3 g((N + 63) / 64, (M + 127) / 128, batch);
    if (gelu)
      gemm_wmma_f16<1, 0><<<g, 256, 0, stream>>>(A, Bm, bias, oh, of, M, N, K, lda, ldb, ldc,
                                                 alpha, bT, zdiv, aOut, aIn, bOut, bIn, cOut, cIn);
    else if (res)
      gemm_wmma_f16<0, 1><<<g, 256, 0, stream>>>(A, Bm, bias, oh, of, M, N, K, lda, ldb, ldc,
                                                 alpha, bT, zdiv, aOut, aIn, bOut, bIn, cOut, cIn);
    else
      gemm_wmma_f16<0, 0><<<g, 256, 0, stream>>>(A, Bm, bias, oh, of, M, N, K, lda, ldb, ldc,
                                                 alpha, bT, zdiv, aOut, aIn, bOut, bIn, cOut, cIn);
  };
  auto conv = [&](const float* s, _Float16* d, long long n) {
    int blocks = (int)((n + 255) / 256); if (blocks > 4096) blocks = 4096;
    convert_f32_f16<<<blocks, 256, 0, stream>>>(s, d, n);
  };

  // ---- 1) weights -> f16 ----
  conv(patchW, wPatch16, szDD);
  conv(Wq, wQ16, DEPTH * szDD); conv(Wk, wK16, DEPTH * szDD);
  conv(Wv, wV16, DEPTH * szDD); conv(Wo, wO16, DEPTH * szDD);
  conv(W1, w116, DEPTH * szDM); conv(W2, w216, DEPTH * szDM);
  conv(headW, wH16, (long long)DIM * NCLS);

  // ---- 2) patchify + patch embedding + cls/pos ----
  {
    long long tot = (long long)MPAT * DIM;
    patchify<<<(int)((tot + 255) / 256), 256, 0, stream>>>(x, xp);
  }
  gemm(xp, wPatch16, patchB, e16, nullptr,
       MPAT, DIM, DIM, DIM, DIM, DIM, 1.0f, 0, 1, 1, 0, 0, 0, 0, 0, 0, 0, 0);
  {
    long long tot = (long long)MTOK * DIM;
    add_cls_pos<<<(int)((tot + 255) / 256), 256, 0, stream>>>(e16, clsTok, posEmb, h32);
  }

  const long long tokStride = (long long)SEQ * DIM;   // 151296 (per image)
  const long long scRow     = (long long)SEQ * SP;    // 44128 (per head score block)

  // ---- 3) transformer layers ----
  for (int i = 0; i < DEPTH; ++i) {
    layernorm_rows<<<MTOK, 256, 0, stream>>>(h32, ln1g + i * DIM, ln1b + i * DIM, y16);
    // Q,K,V projections: [6304,768] x [768,768]
    gemm(y16, wQ16 + (long long)i * szDD, bq + i * DIM, q16, nullptr,
         MTOK, DIM, DIM, DIM, DIM, DIM, 1.0f, 0, 1, 1, 0, 0, 0, 0, 0, 0, 0, 0);
    gemm(y16, wK16 + (long long)i * szDD, bk + i * DIM, k16, nullptr,
         MTOK, DIM, DIM, DIM, DIM, DIM, 1.0f, 0, 1, 1, 0, 0, 0, 0, 0, 0, 0, 0);
    gemm(y16, wV16 + (long long)i * szDD, bv + i * DIM, v16, nullptr,
         MTOK, DIM, DIM, DIM, DIM, DIM, 1.0f, 0, 1, 1, 0, 0, 0, 0, 0, 0, 0, 0);
    // scores[z=b*H+h] = (Q_bh * K_bh^T) / 8 : batched 384x(197x197x64), f32 out (ld 224)
    gemm(q16, k16, nullptr, nullptr, sc32,
         SEQ, SEQ, DK, DIM, DIM, SP, 0.125f, /*bT=*/1, BATCH * HEADS, HEADS,
         tokStride, DK, tokStride, DK, (long long)HEADS * scRow, scRow, 0, 0);
    softmax_rows<<<BATCH * HEADS * SEQ, 256, 0, stream>>>(sc32, p16);
    // att[z] = P_bh * V_bh : batched 384x(197x64x197), f16 out into token-major layout
    gemm(p16, v16, nullptr, att, nullptr,
         SEQ, DK, SEQ, SP, DIM, DIM, 1.0f, 0, BATCH * HEADS, HEADS,
         (long long)HEADS * scRow, scRow, tokStride, DK, tokStride, DK, 0, 0);
    // output projection + residual accumulate into h32
    gemm(att, wO16 + (long long)i * szDD, bo + i * DIM, nullptr, h32,
         MTOK, DIM, DIM, DIM, DIM, DIM, 1.0f, 0, 1, 1, 0, 0, 0, 0, 0, 0, 0, /*res=*/1);
    layernorm_rows<<<MTOK, 256, 0, stream>>>(h32, ln2g + i * DIM, ln2b + i * DIM, y16);
    // MLP up + exact GELU
    gemm(y16, w116 + (long long)i * szDM, b1 + i * MLPD, mlp, nullptr,
         MTOK, MLPD, DIM, DIM, MLPD, MLPD, 1.0f, 0, 1, 1, 0, 0, 0, 0, 0, 0, /*gelu=*/1, 0);
    // MLP down + residual accumulate
    gemm(mlp, w216 + (long long)i * szDM, b2 + i * DIM, nullptr, h32,
         MTOK, DIM, MLPD, MLPD, DIM, DIM, 1.0f, 0, 1, 1, 0, 0, 0, 0, 0, 0, 0, /*res=*/1);
  }

  // ---- 4) final LN + classifier head on cls tokens ----
  layernorm_rows<<<MTOK, 256, 0, stream>>>(h32, lnpg, lnpb, y16);
  // cls rows are rows b*SEQ of y16 -> A with lda = SEQ*DIM picks exactly h[:,0,:]
  gemm(y16, wH16, headB, nullptr, out,
       BATCH, NCLS, DIM, (int)tokStride, NCLS, NCLS, 1.0f, 0, 1, 1, 0, 0, 0, 0, 0, 0, 0, 0);
}
